// DogDetector_74594991997017
// MI455X (gfx1250) — matrix operand order repaired and checked by reading, not compiled, not run on hardware
//
#include <hip/hip_runtime.h>
#include <math.h>

typedef __attribute__((ext_vector_type(16))) _Float16 v16h;
typedef __attribute__((ext_vector_type(8)))  _Float16 v8h;
typedef __attribute__((ext_vector_type(8)))  float    v8f;

#define HWD   14
#define NPIX  196
#define BATCH 32
#define NANCH 1764
#define NKEEP 50
#define ROWH  40   // LDS row stride (halves): 80B = 16B-aligned, conflict-reduced
#define XPSLACK (131072 / 2)  // 128KB tail slack (halves) for branch-free edge reads

#define V16CAT(lo, hi8) __builtin_shufflevector(lo, hi8, 0,1,2,3,4,5,6,7,8,9,10,11,12,13,14,15)

// ---------------------------------------------------------------------------
// Weight repack: W [Cout][Cin][3][3] f32 -> Wr [Cout][9*Cin] f16, k = rs*Cin+ci.
// ---------------------------------------------------------------------------
__global__ void pack_w_f16(const float* __restrict__ W, _Float16* __restrict__ Wr,
                           int Cout, int Cin) {
  int gid = blockIdx.x * blockDim.x + threadIdx.x;
  const int total = Cout * Cin * 9;
  if (gid >= total) return;
  const int K  = Cin * 9;
  const int co = gid / K;
  const int rem = gid - co * K;
  const int rs = rem / Cin;
  const int ci = rem - rs * Cin;
  Wr[gid] = (_Float16)W[((size_t)co * Cin + ci) * 9 + rs];
}

// ---------------------------------------------------------------------------
// Activation pack: X [B,C,14,14] f32 -> Xp [B,256,C] f16 (channel-LAST,
// zero halo). Channel-last makes conv B-staging two b128 loads per thread.
// ---------------------------------------------------------------------------
__global__ void pack_x_f16(const float* __restrict__ X, _Float16* __restrict__ Xp,
                           int C) {
  int gid = blockIdx.x * blockDim.x + threadIdx.x;
  const int total = BATCH * 256 * C;
  if (gid >= total) return;
  const int c   = gid % C;
  const int pos = (gid / C) & 255;
  const int b   = gid / (C * 256);
  const int yp = pos >> 4, xp = pos & 15;
  const int y = yp - 1, x = xp - 1;
  float v = 0.0f;
  if (y >= 0 && y < HWD && x >= 0 && x < HWD)
    v = X[((size_t)b * C + c) * NPIX + y * HWD + x];
  Xp[gid] = (_Float16)v;
}

// ---------------------------------------------------------------------------
// Conv3x3 (pad 1) as implicit GEMM on v_wmma_f32_16x16x32_f16.
// Grid: (4 px-tiles of 64, Cout/32, B). Block: 128 = 4 waves.
// Wave w: 32(co) x 16(px) tile -> 2 WMMAs per 32-wide K chunk (k = rs*Cin+ci).
//   A frags: direct global b128 pairs from packed Wr (no LDS, L2-resident)
//   B tile : branch-free (overrun-tolerant) channel-last gather:
//            2 x global_load_b128 + 2 x ds_store_b128 per thread per chunk.
//   LDS double-buffered -> single barrier per chunk.
// Garbage columns (p >= 196) are computed but never stored.
// ---------------------------------------------------------------------------
__global__ __launch_bounds__(128)
void conv3x3_wmma(const _Float16* __restrict__ Xp,   // [B][256][Cin] padded, +slack
                  const _Float16* __restrict__ Wr,   // [Cout][9*Cin]
                  const float* __restrict__ bias,
                  float* __restrict__ Y,             // [B][Cout][196] f32
                  int Cin, int Cout) {
  const int b       = blockIdx.z;
  const int co_base = blockIdx.y * 32;
  const int n_base  = blockIdx.x * 64;
  const int tid  = threadIdx.x;
  const int lane = tid & 31;
  const int wave = tid >> 5;
  const int Kt = Cin * 9;

  __shared__ __align__(16) _Float16 sB[2][64 * ROWH];

  // fixed staging pixel per thread (p >= 196 reads land in tail slack: harmless)
  const int sn   = tid & 63;
  const int skk0 = (tid >> 6) * 16;      // threads 0-63: kk 0..15, 64-127: 16..31
  const int p    = n_base + sn;
  const int py   = p / HWD;
  const int px   = p - py * HWD;

  // fragment base addresses (CDNA5 16-bit WMMA layouts)
  const int mr = lane & 15;
  const int hi = lane >> 4;
  const _Float16* a0base = Wr + (size_t)(co_base + mr) * Kt + (hi ? 8 : 0);
  const _Float16* a1base = a0base + (size_t)16 * Kt;
  const _Float16* XpB    = Xp + (size_t)b * 256 * Cin;
  _Float16* sBwr = &sB[0][sn * ROWH + skk0];
  const int bplo = (wave * 16 + mr) * ROWH + (hi ? 8 : 0);

  v8f acc0 = {}, acc1 = {};
  int par = 0;

  for (int rs = 0; rs < 9; ++rs) {
    const int r = rs / 3;
    const int s = rs - r * 3;
    const _Float16* xsrc = XpB + (size_t)((py + r) * 16 + (px + s)) * Cin + skk0;
    for (int ci0 = 0; ci0 < Cin; ci0 += 32) {
      // --- stage B: branch-free, fully vectorized ---
      const _Float16* src = xsrc + ci0;
      v8h t0 = *(const v8h*)src;
      v8h t1 = *(const v8h*)(src + 8);
      _Float16* dst = sBwr + par * (64 * ROWH);
      *(v8h*)dst       = t0;
      *(v8h*)(dst + 8) = t1;
      __syncthreads();

      const int k0 = rs * Cin + ci0;
      const _Float16* ap0 = a0base + k0;
      const _Float16* ap1 = a1base + k0;
      __builtin_prefetch(ap0 + 32, 0, 0);         // next A chunk -> global_prefetch
      v8h a0lo = *(const v8h*)(ap0);
      v8h a0hi = *(const v8h*)(ap0 + 16);
      v16h af0 = V16CAT(a0lo, a0hi);
      v8h a1lo = *(const v8h*)(ap1);
      v8h a1hi = *(const v8h*)(ap1 + 16);
      v16h af1 = V16CAT(a1lo, a1hi);
      const _Float16* bpl = &sB[0][0] + par * (64 * ROWH) + bplo;
      v8h blo = *(const v8h*)bpl;
      v8h bhi = *(const v8h*)(bpl + 16);
      v16h bf = V16CAT(blo, bhi);

      acc0 = __builtin_amdgcn_wmma_f32_16x16x32_f16(false, af0, false, bf,
                                                    (short)0, acc0, false, false);
      acc1 = __builtin_amdgcn_wmma_f32_16x16x32_f16(false, af1, false, bf,
                                                    (short)0, acc1, false, false);
      par ^= 1;
    }
  }

  // store: C/D layout: VGPR r -> (M = r + 8*hi, N = mr)
  const int n = n_base + wave * 16 + mr;
  if (n < NPIX) {
    float* yb = Y + (size_t)b * Cout * NPIX + n;
#pragma unroll
    for (int r2 = 0; r2 < 8; ++r2) {
      const int co = co_base + hi * 8 + r2;
      yb[(size_t)co * NPIX]        = acc0[r2] + bias[co];
      yb[(size_t)(co + 16) * NPIX] = acc1[r2] + bias[co + 16];
    }
  }
}

// ---------------------------------------------------------------------------
// Fused erf-GELU + GroupNorm. One block per (group, image); group = 8ch x 196px.
// Writes f32 [B,C,196] (heads) and channel-last zero-haloed f16 [B,256,C]
// (next conv), including the halo cells.
// ---------------------------------------------------------------------------
__global__ __launch_bounds__(256)
void gelu_groupnorm(const float* __restrict__ X, const float* __restrict__ gw,
                    const float* __restrict__ gb, float* __restrict__ Yf,
                    _Float16* __restrict__ Xp, int C, int G) {
  const int b = blockIdx.y;
  const int g = blockIdx.x;
  const int cg = C / G;          // always 8 here
  const int M = cg * NPIX;       // 1568
  __shared__ float vals[8 * NPIX];
  __shared__ float rsum[256], rsq[256];
  const int tid = threadIdx.x;

  float s = 0.0f, q = 0.0f;
  for (int e = tid; e < M; e += 256) {
    const int ec = e / NPIX;
    const int p  = e - ec * NPIX;
    const int c  = g * cg + ec;
    float x  = X[((size_t)b * C + c) * NPIX + p];
    float gv = 0.5f * x * (1.0f + erff(x * 0.7071067811865476f));
    vals[e] = gv;
    s += gv; q += gv * gv;
  }
  // zero the f16 halo cells for this group's channels (channel-last layout)
  for (int e = tid; e < cg * 256; e += 256) {
    const int c   = g * cg + (e >> 8);
    const int pos = e & 255;
    const int yp = pos >> 4, xp = pos & 15;
    if (yp == 0 || yp == 15 || xp == 0 || xp == 15)
      Xp[((size_t)b * 256 + pos) * C + c] = (_Float16)0.0f;
  }
  rsum[tid] = s; rsq[tid] = q;
  __syncthreads();
  for (int st = 128; st > 0; st >>= 1) {
    if (tid < st) { rsum[tid] += rsum[tid + st]; rsq[tid] += rsq[tid + st]; }
    __syncthreads();
  }
  const float mean = rsum[0] / (float)M;
  const float var  = rsq[0] / (float)M - mean * mean;
  const float rstd = rsqrtf(var + 1e-5f);

  for (int e = tid; e < M; e += 256) {
    const int ec = e / NPIX;
    const int p  = e - ec * NPIX;
    const int c  = g * cg + ec;
    const float vv = (vals[e] - mean) * rstd * gw[c] + gb[c];
    Yf[((size_t)b * C + c) * NPIX + p] = vv;
    const int pyy = p / HWD;
    const int pxx = p - pyy * HWD;
    const int pos = (pyy + 1) * 16 + (pxx + 1);
    Xp[((size_t)b * 256 + pos) * C + c] = (_Float16)vv;
  }
}

// ---------------------------------------------------------------------------
// 1x1 heads (K=128) with anchor-layout permute fused in.
// ---------------------------------------------------------------------------
__global__ void bbox_head(const float* __restrict__ X, const float* __restrict__ W,
                          const float* __restrict__ bias, float* __restrict__ bbox) {
  int gid = blockIdx.x * blockDim.x + threadIdx.x;
  if (gid >= BATCH * 36 * NPIX) return;
  const int p  = gid % NPIX;
  const int co = (gid / NPIX) % 36;
  const int b  = gid / (NPIX * 36);
  const float* xp = X + (size_t)b * 128 * NPIX + p;
  const float* wp = W + co * 128;
  float acc = bias[co];
  for (int ci = 0; ci < 128; ++ci) acc += wp[ci] * xp[(size_t)ci * NPIX];
  const int a = co >> 2, c = co & 3;
  bbox[((size_t)b * NANCH + p * 9 + a) * 4 + c] = acc;
}

__global__ void conf_head(const float* __restrict__ X, const float* __restrict__ W,
                          const float* __restrict__ bias, float* __restrict__ conf) {
  int gid = blockIdx.x * blockDim.x + threadIdx.x;
  if (gid >= BATCH * 9 * NPIX) return;
  const int p  = gid % NPIX;
  const int co = (gid / NPIX) % 9;
  const int b  = gid / (NPIX * 9);
  const float* xp = X + (size_t)b * 128 * NPIX + p;
  const float* wp = W + co * 128;
  float acc = bias[co];
  for (int ci = 0; ci < 128; ++ci) acc += wp[ci] * xp[(size_t)ci * NPIX];
  conf[(size_t)b * NANCH + p * 9 + co] = 1.0f / (1.0f + expf(-acc));
}

// ---------------------------------------------------------------------------
// Greedy NMS, one 256-thread block per image, fully in LDS (~53 KB).
// ---------------------------------------------------------------------------
__global__ __launch_bounds__(256)
void nms_kernel(const float* __restrict__ bbox, const float* __restrict__ conf,
                float* __restrict__ out) {
  const int b   = blockIdx.x;
  const int tid = threadIdx.x;
  __shared__ float key[2048];
  __shared__ int   kidx[2048];
  __shared__ float bx1[NANCH], by1[NANCH], bx2[NANCH], by2[NANCH], barea[NANCH];
  __shared__ unsigned char sup[NANCH];
  __shared__ int keepIdx[NKEEP];
  __shared__ int scount;

  const float NEG = -__builtin_inff();
  for (int i = tid; i < 2048; i += 256) {
    float s = (i < NANCH) ? conf[(size_t)b * NANCH + i] : NEG;
    key[i]  = (i < NANCH && s > 0.5f) ? s : NEG;
    kidx[i] = i;
  }
  if (tid == 0) scount = 0;
  __syncthreads();

  // bitonic sort, descending
  for (int k = 2; k <= 2048; k <<= 1) {
    for (int j = k >> 1; j > 0; j >>= 1) {
      for (int i = tid; i < 2048; i += 256) {
        const int ix = i ^ j;
        if (ix > i) {
          float ka = key[i], kb = key[ix];
          const bool sw = ((i & k) == 0) ? (ka < kb) : (ka > kb);
          if (sw) {
            key[i] = kb; key[ix] = ka;
            int t = kidx[i]; kidx[i] = kidx[ix]; kidx[ix] = t;
          }
        }
      }
      __syncthreads();
    }
  }

  // gather clipped boxes in sorted order
  for (int i = tid; i < NANCH; i += 256) {
    const float* bp = bbox + ((size_t)b * NANCH + kidx[i]) * 4;
    float x1 = fminf(fmaxf(bp[0], 0.0f), 1.0f);
    float y1 = fminf(fmaxf(bp[1], 0.0f), 1.0f);
    float x2 = fminf(fmaxf(bp[2], 0.0f), 1.0f);
    float y2 = fminf(fmaxf(bp[3], 0.0f), 1.0f);
    bx1[i] = x1; by1[i] = y1; bx2[i] = x2; by2[i] = y2;
    barea[i] = (x2 - x1) * (y2 - y1);
    sup[i] = 0;
  }

  // greedy loop (uniform control from shared data)
  for (int i = 0; i < NANCH; ++i) {
    __syncthreads();
    if (scount >= NKEEP) break;
    if (key[i] == NEG) break;        // sorted: rest invalid
    if (sup[i]) continue;
    if (tid == 0) { keepIdx[scount] = i; scount++; }
    const float ix1 = bx1[i], iy1 = by1[i], ix2 = bx2[i], iy2 = by2[i], ia = barea[i];
    for (int j = i + 1 + tid; j < NANCH; j += 256) {
      if (!sup[j]) {
        float xx1 = fmaxf(ix1, bx1[j]);
        float yy1 = fmaxf(iy1, by1[j]);
        float xx2 = fminf(ix2, bx2[j]);
        float yy2 = fminf(iy2, by2[j]);
        float inter = fmaxf(xx2 - xx1, 0.0f) * fmaxf(yy2 - yy1, 0.0f);
        float iou = inter / (ia + barea[j] - inter);
        if (!(iou < 0.3f)) sup[j] = 1;   // NaN -> suppressed, like reference
      }
    }
  }
  __syncthreads();

  if (tid < NKEEP) {
    float* op = out + ((size_t)b * NKEEP + tid) * 5;
    if (tid < scount) {
      const int i = keepIdx[tid];
      op[0] = bx1[i]; op[1] = by1[i]; op[2] = bx2[i]; op[3] = by2[i]; op[4] = key[i];
    } else {
      op[0] = 0.0f; op[1] = 0.0f; op[2] = 0.0f; op[3] = 0.0f; op[4] = -1.0f;
    }
  }
}

// ---------------------------------------------------------------------------
extern "C" void kernel_launch(void* const* d_in, const int* in_sizes, int n_in,
                              void* d_out, int out_size, void* d_ws, size_t ws_size,
                              hipStream_t stream) {
  (void)in_sizes; (void)n_in; (void)out_size; (void)ws_size;
  const float* feat = (const float*)d_in[0];
  const float* b1w  = (const float*)d_in[1];  const float* b1b  = (const float*)d_in[2];
  const float* bg1w = (const float*)d_in[3];  const float* bg1b = (const float*)d_in[4];
  const float* b2w  = (const float*)d_in[5];  const float* b2b  = (const float*)d_in[6];
  const float* bg2w = (const float*)d_in[7];  const float* bg2b = (const float*)d_in[8];
  const float* b3w  = (const float*)d_in[9];  const float* b3b  = (const float*)d_in[10];
  const float* bg3w = (const float*)d_in[11]; const float* bg3b = (const float*)d_in[12];
  const float* b4w  = (const float*)d_in[13]; const float* b4b  = (const float*)d_in[14];
  const float* c1w  = (const float*)d_in[15]; const float* c1b  = (const float*)d_in[16];
  const float* cg1w = (const float*)d_in[17]; const float* cg1b = (const float*)d_in[18];
  const float* c2w  = (const float*)d_in[19]; const float* c2b  = (const float*)d_in[20];
  const float* cg2w = (const float*)d_in[21]; const float* cg2b = (const float*)d_in[22];
  const float* c3w  = (const float*)d_in[23]; const float* c3b  = (const float*)d_in[24];

  // ---- workspace carve-out (bytes, 256B aligned slices) ----
  char* ws = (char*)d_ws;
  size_t off = 0;
  auto carve = [&](size_t bytes) { char* p = ws + off; off += (bytes + 255) & ~(size_t)255; return p; };
  _Float16* wr1  = (_Float16*)carve((size_t)256 * 4608 * 2);
  _Float16* wr2  = (_Float16*)carve((size_t)128 * 2304 * 2);
  _Float16* wr3  = (_Float16*)carve((size_t)128 * 1152 * 2);
  _Float16* wrc1 = (_Float16*)carve((size_t)256 * 4608 * 2);
  _Float16* wrc2 = (_Float16*)carve((size_t)128 * 2304 * 2);
  _Float16* xpF  = (_Float16*)carve(((size_t)BATCH * 256 * 512 + XPSLACK) * 2);
  _Float16* xpG  = (_Float16*)carve(((size_t)BATCH * 256 * 256 + XPSLACK) * 2);
  _Float16* xpG2 = (_Float16*)carve(((size_t)BATCH * 256 * 256 + XPSLACK) * 2);
  float* convOut = (float*)carve((size_t)BATCH * 256 * NPIX * 4);
  float* gnF32   = (float*)carve((size_t)BATCH * 256 * NPIX * 4);
  float* bboxb   = (float*)carve((size_t)BATCH * NANCH * 4 * 4);
  float* confb   = (float*)carve((size_t)BATCH * NANCH * 4);

  // ---- one-time per-launch packs (deterministic) ----
  pack_w_f16<<<(256 * 4608 + 255) / 256, 256, 0, stream>>>(b1w, wr1, 256, 512);
  pack_w_f16<<<(128 * 2304 + 255) / 256, 256, 0, stream>>>(b2w, wr2, 128, 256);
  pack_w_f16<<<(128 * 1152 + 255) / 256, 256, 0, stream>>>(b3w, wr3, 128, 128);
  pack_w_f16<<<(256 * 4608 + 255) / 256, 256, 0, stream>>>(c1w, wrc1, 256, 512);
  pack_w_f16<<<(128 * 2304 + 255) / 256, 256, 0, stream>>>(c2w, wrc2, 128, 256);
  pack_x_f16<<<(BATCH * 512 * 256 + 255) / 256, 256, 0, stream>>>(feat, xpF, 512);

  // ---- bbox branch ----
  conv3x3_wmma<<<dim3(4, 8, BATCH), 128, 0, stream>>>(xpF, wr1, b1b, convOut, 512, 256);
  gelu_groupnorm<<<dim3(32, BATCH), 256, 0, stream>>>(convOut, bg1w, bg1b, gnF32, xpG, 256, 32);
  conv3x3_wmma<<<dim3(4, 4, BATCH), 128, 0, stream>>>(xpG, wr2, b2b, convOut, 256, 128);
  gelu_groupnorm<<<dim3(16, BATCH), 256, 0, stream>>>(convOut, bg2w, bg2b, gnF32, xpG2, 128, 16);
  conv3x3_wmma<<<dim3(4, 4, BATCH), 128, 0, stream>>>(xpG2, wr3, b3b, convOut, 128, 128);
  gelu_groupnorm<<<dim3(16, BATCH), 256, 0, stream>>>(convOut, bg3w, bg3b, gnF32, xpG, 128, 16);
  bbox_head<<<(BATCH * 36 * NPIX + 255) / 256, 256, 0, stream>>>(gnF32, b4w, b4b, bboxb);

  // ---- classification branch ----
  conv3x3_wmma<<<dim3(4, 8, BATCH), 128, 0, stream>>>(xpF, wrc1, c1b, convOut, 512, 256);
  gelu_groupnorm<<<dim3(32, BATCH), 256, 0, stream>>>(convOut, cg1w, cg1b, gnF32, xpG, 256, 32);
  conv3x3_wmma<<<dim3(4, 4, BATCH), 128, 0, stream>>>(xpG, wrc2, c2b, convOut, 256, 128);
  gelu_groupnorm<<<dim3(16, BATCH), 256, 0, stream>>>(convOut, cg2w, cg2b, gnF32, xpG2, 128, 16);
  conf_head<<<(BATCH * 9 * NPIX + 255) / 256, 256, 0, stream>>>(gnF32, c3w, c3b, confb);

  // ---- NMS ----
  nms_kernel<<<BATCH, 256, 0, stream>>>(bboxb, confb, (float*)d_out);
}
